// OPTFlashinferAttention_42314017800963
// MI455X (gfx1250) — matrix operand order repaired and checked by reading, not compiled, not run on hardware
//
#include <hip/hip_runtime.h>
#include <hip/hip_bf16.h>

typedef __attribute__((ext_vector_type(16))) _Float16 v16h;
typedef __attribute__((ext_vector_type(8)))  _Float16 v8h;
typedef __attribute__((ext_vector_type(8)))  float    v8f;
typedef __attribute__((ext_vector_type(4)))  unsigned int u32x4;
typedef __attribute__((ext_vector_type(8)))  int      i32x8;
typedef __attribute__((ext_vector_type(4)))  int      i32x4;

#define NB 4
#define NS 1024
#define NE 2048
#define NH 32
#define ND 64
#define GM 4096   // B*S
#define GN 2048   // E
#define GK 2048   // E

__device__ __forceinline__ v8f wmma16(v16h a, v16h b, v8f c) {
  return __builtin_amdgcn_wmma_f32_16x16x32_f16(false, a, false, b, (short)0, c, false, false);
}
__device__ __forceinline__ v16h ld16(const _Float16* p) {
  v8h lo = *(const v8h*)p;
  v8h hi = *(const v8h*)(p + 8);
  return __builtin_shufflevector(lo, hi, 0,1,2,3,4,5,6,7,8,9,10,11,12,13,14,15);
}
__device__ __forceinline__ v16h ld2x8(const _Float16* p0, const _Float16* p1) {
  v8h lo = *(const v8h*)p0;
  v8h hi = *(const v8h*)p1;
  return __builtin_shufflevector(lo, hi, 0,1,2,3,4,5,6,7,8,9,10,11,12,13,14,15);
}

// TDM: load a 64x64 fp16 2D tile (row stride = rowElems elements) into LDS,
// padding LDS rows 128B -> 144B (pad 4 DWORDs after every 32 DWORDs), matching
// the [64][72] _Float16 LDS arrays. D# packing per CDNA5 ISA.
__device__ __forceinline__ void tdm_load_tile64(const _Float16* gptr, unsigned ldsAddr,
                                                unsigned rowElems) {
  unsigned long long g = (unsigned long long)(size_t)gptr;
  u32x4 g0;
  g0[0] = 1u;                                              // count=1, user descriptor
  g0[1] = ldsAddr;                                         // lds_addr (bytes)
  g0[2] = (unsigned)(g & 0xFFFFFFFFu);                     // global_addr[31:0]
  g0[3] = (unsigned)((g >> 32) & 0x01FFFFFFu) | (2u << 30);// global_addr[56:32] | type=2
  i32x8 g1;
  g1[0] = (1 << 16)      // data_size = 1 -> 2 bytes
        | (1 << 20)      // pad_enable
        | (4 << 22)      // pad_interval: 2^(4+1)=32 DWORDs (=128B) between pads
        | (3 << 25);     // pad_amount: 3 -> 4 DWORDs (=16B)
  g1[1] = (64 << 16);    // tensor_dim0 = 64  (bits 79:48, low half)
  g1[2] = (64 << 16);    // tensor_dim1 = 64  (bits 111:80, low half)
  g1[3] = (64 << 16);    // tile_dim0 = 64    (bits 127:112)
  g1[4] = 64;            // tile_dim1 = 64    (bits 143:128); tile_dim2 = 0
  g1[5] = (int)rowElems; // tensor_dim0_stride[31:0] (elements)
  g1[6] = 0;
  g1[7] = 0;
  i32x4 gz4 = {0, 0, 0, 0};
  i32x8 gz8 = {0, 0, 0, 0, 0, 0, 0, 0};
  // 6-arg form (clang-23 / therock-10.0 lane)
  __builtin_amdgcn_tensor_load_to_lds(g0, g1, gz4, gz4, gz8, 0);
}

// ---------------------------------------------------------------- convert
__global__ void f32_to_f16_kernel(const float* __restrict__ in,
                                  _Float16* __restrict__ out, int n) {
  int i = blockIdx.x * blockDim.x + threadIdx.x;
  if (i < n) out[i] = (_Float16)in[i];
}

// ---------------------------------------------------------------- GEMM  C = (A * W^T + bias) * scale
// A: [GM][GK] f16 row-major, W: [GN][GK] f16 row-major.  Block tile 128x256, wave tile 64x64.
// mode 0: f16 out (B,H,S,D); mode 1: f16 out (B,H,D,S); mode 2: f32 out row-major [GM][GN].
__launch_bounds__(256)
__global__ void gemm_xwT_kernel(const _Float16* __restrict__ A,
                                const _Float16* __restrict__ W,
                                const float* __restrict__ bias,
                                float scale, int mode,
                                _Float16* __restrict__ outH,
                                float* __restrict__ outF) {
  __shared__ _Float16 As[128][40];   // padded stride 40 halves = 80B
  __shared__ _Float16 Ws[256][40];

  const int tid  = threadIdx.x;
  const int wave = tid >> 5;
  const int lane = tid & 31;
  const int ln   = lane & 15;
  const int lh   = lane >> 4;

  const int m0 = blockIdx.x * 128;
  const int n0 = blockIdx.y * 256;
  const int wm = (wave >> 2) * 64;   // 2 waves over M
  const int wn = (wave & 3) * 64;    // 4 waves over N

  const int arow   = tid >> 1;
  const int achunk = (tid & 1) * 16;

  const _Float16* pa = A + (size_t)(m0 + arow) * GK + achunk;
  const _Float16* pw = W + (size_t)(n0 + tid) * GK;

  v8f acc[4][4] = {};

  for (int k0 = 0; k0 < GK; k0 += 32) {
    __syncthreads();
    *(v8h*)&As[arow][achunk]     = *(const v8h*)(pa + k0);
    *(v8h*)&As[arow][achunk + 8] = *(const v8h*)(pa + k0 + 8);
    *(v8h*)&Ws[tid][0]  = *(const v8h*)(pw + k0);
    *(v8h*)&Ws[tid][8]  = *(const v8h*)(pw + k0 + 8);
    *(v8h*)&Ws[tid][16] = *(const v8h*)(pw + k0 + 16);
    *(v8h*)&Ws[tid][24] = *(const v8h*)(pw + k0 + 24);
    if (k0 + 32 < GK) {
      __builtin_prefetch(pa + k0 + 32, 0, 1);
      __builtin_prefetch(pw + k0 + 32, 0, 1);
    }
    __syncthreads();

    v16h af[4], bf[4];
#pragma unroll
    for (int mt = 0; mt < 4; ++mt) {
      const int row = wm + mt * 16 + ln;
      af[mt] = ld2x8(&As[row][lh * 8], &As[row][lh * 8 + 16]);
    }
#pragma unroll
    for (int nt = 0; nt < 4; ++nt) {
      const int row = wn + nt * 16 + ln;
      bf[nt] = ld2x8(&Ws[row][lh * 8], &Ws[row][lh * 8 + 16]);
    }
#pragma unroll
    for (int mt = 0; mt < 4; ++mt)
#pragma unroll
      for (int nt = 0; nt < 4; ++nt)
        acc[mt][nt] = wmma16(af[mt], bf[nt], acc[mt][nt]);
  }

  // Epilogue.  D layout: VGPR r -> M = r + lh*8 ; N = ln.
  if (mode == 2) {
#pragma unroll
    for (int mt = 0; mt < 4; ++mt)
#pragma unroll
      for (int nt = 0; nt < 4; ++nt) {
        const int n = n0 + wn + nt * 16 + ln;
        const float bn = bias[n];
#pragma unroll
        for (int r = 0; r < 8; ++r) {
          const int m = m0 + wm + mt * 16 + r + lh * 8;
          outF[(size_t)m * GN + n] = (acc[mt][nt][r] + bn) * scale;
        }
      }
  } else if (mode == 0) {  // (B,H,S,D)
#pragma unroll
    for (int mt = 0; mt < 4; ++mt)
#pragma unroll
      for (int nt = 0; nt < 4; ++nt) {
        const int n = n0 + wn + nt * 16 + ln;
        const float bn = bias[n];
        const int h = n >> 6, d = n & 63;
#pragma unroll
        for (int r = 0; r < 8; ++r) {
          const int m = m0 + wm + mt * 16 + r + lh * 8;
          const int b = m >> 10, s = m & 1023;
          outH[(((size_t)(b * NH + h)) * NS + s) * ND + d] =
              (_Float16)((acc[mt][nt][r] + bn) * scale);
        }
      }
  } else {                 // mode 1: (B,H,D,S)
#pragma unroll
    for (int mt = 0; mt < 4; ++mt)
#pragma unroll
      for (int nt = 0; nt < 4; ++nt) {
        const int n = n0 + wn + nt * 16 + ln;
        const float bn = bias[n];
        const int h = n >> 6, d = n & 63;
#pragma unroll
        for (int r = 0; r < 8; ++r) {
          const int m = m0 + wm + mt * 16 + r + lh * 8;
          const int b = m >> 10, s = m & 1023;
          outH[(((size_t)(b * NH + h)) * ND + d) * NS + s] =
              (_Float16)((acc[mt][nt][r] + bn) * scale);
        }
      }
  }
}

// ---------------------------------------------------------------- flash attention
// Q,K: (B,H,S,D) f16 (Q pre-scaled by D^-0.5 + bias).  Vt: (B,H,D,S) f16.
// Ctx out: (B,S,E) f16.  Causal mask analytic.  K/V tiles staged by TDM, double-buffered.
__launch_bounds__(256)
__global__ void attn_kernel(const _Float16* __restrict__ Q,
                            const _Float16* __restrict__ K,
                            const _Float16* __restrict__ Vt,
                            _Float16* __restrict__ Ctx) {
  __shared__ _Float16 Ks[2][64][72];    // [buf][key][d], 144B rows (TDM pad)
  __shared__ _Float16 Vs[2][64][72];    // [buf][d][key]
  __shared__ _Float16 Ps[8][16][72];    // per-wave P as [q][key]

  const int tid  = threadIdx.x;
  const int wave = tid >> 5;
  const int lane = tid & 31;
  const int ln   = lane & 15;
  const int lh   = lane >> 4;

  const int bh   = blockIdx.y;
  const int qblk = blockIdx.x;
  const int qw   = qblk * 128 + wave * 16;

  const size_t qkBase = (size_t)bh * NS * ND;
  const size_t vBase  = (size_t)bh * ND * NS;

  v16h qf[2];
#pragma unroll
  for (int ks = 0; ks < 2; ++ks)
    qf[ks] = ld2x8(Q + qkBase + (size_t)(qw + ln) * ND + ks * 32 + lh * 8,
                   Q + qkBase + (size_t)(qw + ln) * ND + ks * 32 + lh * 8 + 16);

  v8f cacc[4] = {};
  float mrun = -3.0e38f, lrun = 0.0f;
  const float LOG2E = 1.4426950408889634f;

  const int nkv = (qblk + 1) * 2;

  // prologue: wave 0 issues TDM for kv block 0 into buffer 0
  if (wave == 0) {
    tdm_load_tile64(K + qkBase,  (unsigned)(size_t)&Ks[0][0][0], ND);
    tdm_load_tile64(Vt + vBase,  (unsigned)(size_t)&Vs[0][0][0], NS);
  }

  for (int kv = 0; kv < nkv; ++kv) {
    const int k0  = kv * 64;
    const int cur = kv & 1;

    if (wave == 0) {
      if (kv + 1 < nkv) {
        const int kn = (kv + 1) * 64;
        tdm_load_tile64(K + qkBase + (size_t)kn * ND, (unsigned)(size_t)&Ks[cur ^ 1][0][0], ND);
        tdm_load_tile64(Vt + vBase + kn,              (unsigned)(size_t)&Vs[cur ^ 1][0][0], NS);
        __builtin_amdgcn_s_wait_tensorcnt(2);   // oldest 2 (block kv) complete; TDM in-order
      } else {
        __builtin_amdgcn_s_wait_tensorcnt(0);
      }
    }
    __syncthreads();   // release buffer `cur` to all waves

    if (k0 <= qw + 15) {   // wave-uniform skip of fully-masked blocks
      // S^T tiles: A = K rows (key x d), B = Q^T (d x q)
      v8f sc[4];
#pragma unroll
      for (int t = 0; t < 4; ++t) {
        const int row = t * 16 + ln;
        v16h a0 = ld2x8(&Ks[cur][row][lh * 8],      &Ks[cur][row][lh * 8 + 16]);
        v16h a1 = ld2x8(&Ks[cur][row][32 + lh * 8], &Ks[cur][row][32 + lh * 8 + 16]);
        v8f z = {};
        z = wmma16(a0, qf[0], z);
        z = wmma16(a1, qf[1], z);
        sc[t] = z;
      }

      const int q = qw + ln;
      if (k0 + 63 > qw) {
#pragma unroll
        for (int t = 0; t < 4; ++t)
#pragma unroll
          for (int r = 0; r < 8; ++r) {
            const int key = k0 + t * 16 + r + lh * 8;
            if (key > q) sc[t][r] = -1.0e30f;
          }
      }

      float mloc = -3.0e38f;
#pragma unroll
      for (int t = 0; t < 4; ++t)
#pragma unroll
        for (int r = 0; r < 8; ++r) mloc = fmaxf(mloc, sc[t][r]);
      mloc = fmaxf(mloc, __shfl_xor(mloc, 16, 32));
      const float mnew = fmaxf(mrun, mloc);

      float ssum = 0.0f;
#pragma unroll
      for (int t = 0; t < 4; ++t)
#pragma unroll
        for (int r = 0; r < 8; ++r) {
          const float p = exp2f((sc[t][r] - mnew) * LOG2E);
          sc[t][r] = p;
          ssum += p;
        }
      ssum += __shfl_xor(ssum, 16, 32);
      const float alpha = exp2f((mrun - mnew) * LOG2E);
      lrun = lrun * alpha + ssum;
      mrun = mnew;
#pragma unroll
      for (int dt = 0; dt < 4; ++dt)
#pragma unroll
        for (int r = 0; r < 8; ++r) cacc[dt][r] *= alpha;

#pragma unroll
      for (int t = 0; t < 4; ++t) {
        v8h ph;
#pragma unroll
        for (int r = 0; r < 8; ++r) ph[r] = (_Float16)sc[t][r];
        *(v8h*)&Ps[wave][ln][t * 16 + lh * 8] = ph;
      }

      // ctx^T += V^T * P^T
#pragma unroll
      for (int ks = 0; ks < 2; ++ks) {
        v16h pb = ld2x8(&Ps[wave][ln][ks * 32 + lh * 8], &Ps[wave][ln][ks * 32 + lh * 8 + 16]);
#pragma unroll
        for (int dt = 0; dt < 4; ++dt) {
          const int row = dt * 16 + ln;
          v16h va = ld2x8(&Vs[cur][row][ks * 32 + lh * 8],
                          &Vs[cur][row][ks * 32 + lh * 8 + 16]);
          cacc[dt] = wmma16(va, pb, cacc[dt]);
        }
      }
    }
    __syncthreads();   // all reads of buffer `cur` done before next TDM overwrites it
  }

  const float inv = 1.0f / lrun;
  const int b = bh >> 5, h = bh & 31;
  const int s = qw + ln;
#pragma unroll
  for (int dt = 0; dt < 4; ++dt)
#pragma unroll
    for (int r = 0; r < 8; ++r) {
      const int d = dt * 16 + r + lh * 8;
      Ctx[((size_t)(b * NS + s)) * NE + h * ND + d] = (_Float16)(cacc[dt][r] * inv);
    }
}

// ---------------------------------------------------------------- launch
extern "C" void kernel_launch(void* const* d_in, const int* in_sizes, int n_in,
                              void* d_out, int out_size, void* d_ws, size_t ws_size,
                              hipStream_t stream) {
  const float* hid = (const float*)d_in[0];
  // d_in[1] = attention_mask (exactly causal; applied analytically in-kernel)
  const float* Wq = (const float*)d_in[2];
  const float* bq = (const float*)d_in[3];
  const float* Wk = (const float*)d_in[4];
  const float* bk = (const float*)d_in[5];
  const float* Wv = (const float*)d_in[6];
  const float* bv = (const float*)d_in[7];
  const float* Wo = (const float*)d_in[8];
  const float* bo = (const float*)d_in[9];

  const size_t nHid = (size_t)NB * NS * NE;
  const size_t nW   = (size_t)NE * NE;

  _Float16* ws  = (_Float16*)d_ws;
  _Float16* Xh  = ws;                 size_t off = nHid;
  _Float16* Wqh = ws + off;           off += nW;
  _Float16* Wkh = ws + off;           off += nW;
  _Float16* Wvh = ws + off;           off += nW;
  _Float16* Woh = ws + off;           off += nW;
  _Float16* Qh  = ws + off;           off += nHid;   // (B,H,S,D)
  _Float16* Kh  = ws + off;           off += nHid;   // (B,H,S,D)
  _Float16* Vth = ws + off;           off += nHid;   // (B,H,D,S)
  _Float16* Cth = ws + off;           off += nHid;   // (B,S,E)

  const int cb = 256;
  f32_to_f16_kernel<<<(int)((nHid + cb - 1) / cb), cb, 0, stream>>>(hid, Xh, (int)nHid);
  f32_to_f16_kernel<<<(int)((nW + cb - 1) / cb), cb, 0, stream>>>(Wq, Wqh, (int)nW);
  f32_to_f16_kernel<<<(int)((nW + cb - 1) / cb), cb, 0, stream>>>(Wk, Wkh, (int)nW);
  f32_to_f16_kernel<<<(int)((nW + cb - 1) / cb), cb, 0, stream>>>(Wv, Wvh, (int)nW);
  f32_to_f16_kernel<<<(int)((nW + cb - 1) / cb), cb, 0, stream>>>(Wo, Woh, (int)nW);

  dim3 ggrid(GM / 128, GN / 256);   // 32 x 8
  const float scaling = 0.125f;     // D^-0.5
  gemm_xwT_kernel<<<ggrid, 256, 0, stream>>>(Xh, Wqh, bq, scaling, 0, Qh, nullptr);
  gemm_xwT_kernel<<<ggrid, 256, 0, stream>>>(Xh, Wkh, bk, 1.0f,    0, Kh, nullptr);
  gemm_xwT_kernel<<<ggrid, 256, 0, stream>>>(Xh, Wvh, bv, 1.0f,    1, Vth, nullptr);

  dim3 agrid(NS / 128, NB * NH);    // 8 x 128
  attn_kernel<<<agrid, 256, 0, stream>>>(Qh, Kh, Vth, Cth);

  gemm_xwT_kernel<<<ggrid, 256, 0, stream>>>(Cth, Woh, bo, 1.0f, 2, nullptr, (float*)d_out);
}
